// SelfAttention_90065464197223
// MI455X (gfx1250) — matrix operand order repaired and checked by reading, not compiled, not run on hardware
//
#include <hip/hip_runtime.h>

#define BATCH 8
#define CCH   256
#define NPIX  4096   // 64*64

typedef _Float16 half_t;
typedef __attribute__((ext_vector_type(16))) _Float16 v16h;
typedef __attribute__((ext_vector_type(8)))  _Float16 v8h;
typedef __attribute__((ext_vector_type(8)))  float    v8f;

static __device__ __forceinline__ v16h join8(v8h lo, v8h hi) {
  v16h r;
#pragma unroll
  for (int i = 0; i < 8; ++i) { r[i] = lo[i]; r[i + 8] = hi[i]; }
  return r;
}

static __device__ __forceinline__ v8f zero8() {
  v8f z;
#pragma unroll
  for (int i = 0; i < 8; ++i) z[i] = 0.0f;
  return z;
}

// CDNA5 async global->LDS copy, 16 bytes per active lane (ASYNCcnt-tracked).
// lds_off: byte offset within the workgroup LDS allocation (low 32 bits of a
// generic pointer to __shared__ memory, per the LDS aperture rule).
static __device__ __forceinline__ void async_ld16(unsigned lds_off, const void* gsrc) {
  asm volatile("global_load_async_to_lds_b128 %0, %1, off"
               :: "v"(lds_off), "v"(gsrc) : "memory");
}

static __device__ __forceinline__ unsigned lds_off_of(const void* p) {
  return (unsigned)(size_t)p;   // LDS aperture: addr[31:0] == LDS byte offset
}

// ---------------------------------------------------------------------------
// Kernel 0: weights fp32 -> f16 (one shot; keeps cvt out of the GEMM loop)
// ---------------------------------------------------------------------------
__global__ __launch_bounds__(256) void k_wconv(const float* __restrict__ Wq,
                                               const float* __restrict__ Wk,
                                               const float* __restrict__ Wv,
                                               half_t* __restrict__ Wh) {
  const int i = blockIdx.x * 256 + threadIdx.x;      // 65536 elements per matrix
  Wh[i]               = (half_t)Wq[i];
  Wh[i + 65536]       = (half_t)Wk[i];
  Wh[i + 2 * 65536]   = (half_t)Wv[i];
}

// ---------------------------------------------------------------------------
// Kernel 1: x (B,C,N) fp32  ->  xb (B,N,C) f16   (transpose + downconvert)
// ---------------------------------------------------------------------------
__global__ __launch_bounds__(256) void k_transpose(const float* __restrict__ x,
                                                   half_t* __restrict__ xb) {
  const int bc = blockIdx.x;              // b*C + c
  const int b  = bc >> 8;
  const int c  = bc & 255;
  const float* src = x + (size_t)bc * NPIX;
#pragma unroll
  for (int i = 0; i < NPIX / 256; ++i) {
    const int n = threadIdx.x + i * 256;
    xb[((size_t)b * NPIX + n) * CCH + c] = (half_t)src[n];
  }
}

// ---------------------------------------------------------------------------
// Kernel 2: QKV projection.  One wave computes a 16(n) x 16(o) tile of all
// three outputs, sharing the A (xb) fragment across Q/K/V WMMAs.
//   Qt,Kt stored (B,N,C) f16 ; V stored (B,C,N) f16
// ---------------------------------------------------------------------------
__global__ __launch_bounds__(256) void k_qkv(const half_t* __restrict__ xb,
                                             const half_t* __restrict__ Wh,
                                             const float* __restrict__ bq,
                                             const float* __restrict__ bk,
                                             const float* __restrict__ bv,
                                             half_t* __restrict__ Qt, half_t* __restrict__ Kt,
                                             half_t* __restrict__ Vn) {
  const int lane   = threadIdx.x & 31;
  const int laneLo = lane & 15;
  const int laneHi = lane >> 4;
  const int wave   = blockIdx.x * 8 + (threadIdx.x >> 5);
  const int b      = wave >> 12;           // 4096 tiles per batch
  const int rem    = wave & 4095;
  const int nT     = rem >> 4;
  const int oT     = rem & 15;
  const int n0 = nT * 16, o0 = oT * 16;

  const half_t* arow = xb + ((size_t)b * NPIX + n0 + laneLo) * CCH;
  const half_t* wqr  = Wh +               (size_t)(o0 + laneLo) * CCH;
  const half_t* wkr  = Wh +     65536 +   (size_t)(o0 + laneLo) * CCH;
  const half_t* wvr  = Wh + 2 * 65536 +   (size_t)(o0 + laneLo) * CCH;

  v8f cq = zero8(), ck = zero8(), cv = zero8();

#pragma unroll
  for (int kk = 0; kk < CCH / 32; ++kk) {
    const int cb = kk * 32;
    // A fragment: row = laneLo, K = cb + laneHi*16 + e
    v16h a = join8(*(const v8h*)(arow + cb + laneHi * 16),
                   *(const v8h*)(arow + cb + laneHi * 16 + 8));
    // B fragments: col = laneLo, K = cb + laneHi*16 + e  (contiguous f16 row)
    const half_t* pq = wqr + cb + laneHi * 16;
    const half_t* pk = wkr + cb + laneHi * 16;
    const half_t* pv = wvr + cb + laneHi * 16;
    v16h bqv = join8(*(const v8h*)pq, *(const v8h*)(pq + 8));
    v16h bkv = join8(*(const v8h*)pk, *(const v8h*)(pk + 8));
    v16h bvv = join8(*(const v8h*)pv, *(const v8h*)(pv + 8));
    cq = __builtin_amdgcn_wmma_f32_16x16x32_f16(false, a, false, bqv, (short)0, cq, false, false);
    ck = __builtin_amdgcn_wmma_f32_16x16x32_f16(false, a, false, bkv, (short)0, ck, false, false);
    cv = __builtin_amdgcn_wmma_f32_16x16x32_f16(false, a, false, bvv, (short)0, cv, false, false);
  }

  const float biasq = bq[o0 + laneLo];
  const float biask = bk[o0 + laneLo];
  const float biasv = bv[o0 + laneLo];

  // Qt/Kt: (B,N,C) — D layout: lane = col o, VGPR r = row r + 8*laneHi
#pragma unroll
  for (int r = 0; r < 8; ++r) {
    const int m = r + 8 * laneHi;
    const size_t idx = ((size_t)b * NPIX + n0 + m) * CCH + o0 + laneLo;
    Qt[idx] = (half_t)(cq[r] + biasq);
    Kt[idx] = (half_t)(ck[r] + biask);
  }
  // V: (B,C,N) — per lane 8 consecutive n values -> single 16B store
  v8h vh;
#pragma unroll
  for (int r = 0; r < 8; ++r) vh[r] = (half_t)(cv[r] + biasv);
  *(v8h*)(Vn + ((size_t)b * CCH + o0 + laneLo) * NPIX + n0 + 8 * laneHi) = vh;
}

// ---------------------------------------------------------------------------
// Kernel 3: flash attention.  8 waves/block share one batch; K/V j-tiles are
// double-buffered in LDS via async global->LDS DMA (ASYNCcnt) so tile j+1's
// copy overlaps tile j's WMMA compute.  Each wave owns one 16-row query tile:
// score WMMAs + online softmax + PV WMMAs, all B-fragments fed from LDS.
// ---------------------------------------------------------------------------
__global__ __launch_bounds__(256) void k_attn(const half_t* __restrict__ Qt,
                                              const half_t* __restrict__ Kt,
                                              const half_t* __restrict__ Vn,
                                              const float* __restrict__ x,
                                              const float* __restrict__ gamma,
                                              float* __restrict__ out) {
  __shared__ half_t kls[2][32 * CCH];     // 2 x 16 KB: K tile, [j'][c]
  __shared__ half_t vls[2][CCH * 32];     // 2 x 16 KB: V tile, [c][j']
  __shared__ half_t ptile[8][16 * 32];    //     8 KB: per-wave P staging

  const int tid    = threadIdx.x;
  const int lane   = tid & 31;
  const int laneLo = lane & 15;
  const int laneHi = lane >> 4;
  const int wv     = tid >> 5;
  const int wave   = blockIdx.x * 8 + wv;
  const int b      = wave >> 8;
  const int iT     = wave & 255;
  const int i0     = iT * 16;

  const half_t* kbase = Kt + (size_t)b * NPIX * CCH;   // (N,C)
  const half_t* vbase = Vn + (size_t)b * CCH * NPIX;   // (C,N)
  half_t* pt = ptile[wv];
  const unsigned kls_off = lds_off_of(kls);
  const unsigned vls_off = lds_off_of(vls);

  // issue this thread's 1/256 share of the K (32x256) + V (256x32) tile DMA
  auto stage = [&](int j0, int buf) {
    const char* ksrc = (const char*)(kbase + (size_t)j0 * CCH);  // contiguous 16 KB
    const unsigned koff = kls_off + (unsigned)buf * (32 * CCH * 2);
#pragma unroll
    for (int i = 0; i < 4; ++i)
      async_ld16(koff + tid * 16 + i * 4096, ksrc + tid * 16 + i * 4096);
    const char* vsrc = (const char*)vbase + (size_t)j0 * 2;      // 256 rows x 64 B
    const unsigned voff = vls_off + (unsigned)buf * (CCH * 32 * 2);
    const int bo = (tid & 3) * 16;
#pragma unroll
    for (int i = 0; i < 4; ++i) {
      const int c = (tid >> 2) + i * 64;
      async_ld16(voff + c * 64 + bo, vsrc + (size_t)c * (NPIX * 2) + bo);
    }
  };

  // Q A-fragments for the full C=256 reduction, resident in VGPRs
  v16h qa[8];
  const half_t* qrow = Qt + ((size_t)b * NPIX + i0 + laneLo) * CCH;
#pragma unroll
  for (int kk = 0; kk < 8; ++kk) {
    qa[kk] = join8(*(const v8h*)(qrow + kk * 32 + laneHi * 16),
                   *(const v8h*)(qrow + kk * 32 + laneHi * 16 + 8));
  }

  float mrow[8], lrow[8];
#pragma unroll
  for (int r = 0; r < 8; ++r) { mrow[r] = -3.0e38f; lrow[r] = 0.0f; }
  v8f acc[16];
#pragma unroll
  for (int t = 0; t < 16; ++t) acc[t] = zero8();

  // prologue: stage tile 0 into buffer 0
  stage(0, 0);
  asm volatile("s_wait_asynccnt 0" ::: "memory");
  __syncthreads();

  for (int jt = 0; jt < NPIX / 32; ++jt) {
    const int j0  = jt * 32;
    const int cur = jt & 1;

    // overlap: issue next tile's DMA into the other buffer (no wait yet)
    if (jt + 1 < NPIX / 32) {
      stage(j0 + 32, cur ^ 1);
      __builtin_prefetch((const void*)(kbase + (size_t)(j0 + 64) * CCH), 0, 1);
    }

    // ---- scores: S = Q^T K for two 16-col sub-tiles (B-frags from LDS) ----
    const half_t* kr0 = kls[cur] + (size_t)laneLo * CCH;
    const half_t* kr1 = kls[cur] + (size_t)(16 + laneLo) * CCH;
    v8f s0 = zero8(), s1 = zero8();
#pragma unroll
    for (int kk = 0; kk < 8; ++kk) {
      const int cb = kk * 32;
      v16h b0 = join8(*(const v8h*)(kr0 + cb + laneHi * 16),
                      *(const v8h*)(kr0 + cb + laneHi * 16 + 8));
      v16h b1 = join8(*(const v8h*)(kr1 + cb + laneHi * 16),
                      *(const v8h*)(kr1 + cb + laneHi * 16 + 8));
      s0 = __builtin_amdgcn_wmma_f32_16x16x32_f16(false, qa[kk], false, b0, (short)0, s0, false, false);
      s1 = __builtin_amdgcn_wmma_f32_16x16x32_f16(false, qa[kk], false, b1, (short)0, s1, false, false);
    }

    // ---- online softmax (row r maps to M = r + 8*laneHi, same as acc) ----
    float p0[8], p1[8];
#pragma unroll
    for (int r = 0; r < 8; ++r) {
      float mx = fmaxf(s0[r], s1[r]);
#pragma unroll
      for (int off = 1; off < 16; off <<= 1)
        mx = fmaxf(mx, __shfl_xor(mx, off, 32));
      const float mnew  = fmaxf(mrow[r], mx);
      const float scale = __expf(mrow[r] - mnew);
      mrow[r] = mnew;
      const float e0 = __expf(s0[r] - mnew);
      const float e1 = __expf(s1[r] - mnew);
      float sum = e0 + e1;
#pragma unroll
      for (int off = 1; off < 16; off <<= 1)
        sum += __shfl_xor(sum, off, 32);
      lrow[r] = lrow[r] * scale + sum;
      p0[r] = e0;
      p1[r] = e1;
#pragma unroll
      for (int t = 0; t < 16; ++t) acc[t][r] *= scale;
    }

    // ---- P: D-layout -> LDS -> A-layout fragment (per-wave region) ----
#pragma unroll
    for (int r = 0; r < 8; ++r) {
      const int row = r + 8 * laneHi;
      pt[row * 32 + laneLo]      = (half_t)p0[r];
      pt[row * 32 + 16 + laneLo] = (half_t)p1[r];
    }
    asm volatile("s_wait_dscnt 0" ::: "memory");
    v16h pa = join8(*(const v8h*)(pt + laneLo * 32 + laneHi * 16),
                    *(const v8h*)(pt + laneLo * 32 + laneHi * 16 + 8));
    asm volatile("" ::: "memory");

    // ---- O += P * V^T over all 16 channel tiles (B-frags from LDS) ----
#pragma unroll
    for (int t = 0; t < 16; ++t) {
      const half_t* vr = vls[cur] + (size_t)(t * 16 + laneLo) * 32 + laneHi * 16;
      v16h bv2 = join8(*(const v8h*)vr, *(const v8h*)(vr + 8));
      acc[t] = __builtin_amdgcn_wmma_f32_16x16x32_f16(false, pa, false, bv2, (short)0, acc[t], false, false);
    }

    // epilogue of the step: next tile's DMA done (own share), then block-wide
    // barrier => (a) next buffer visible to all waves, (b) everyone finished
    // reading buf[cur] before it is overwritten at step jt+1.
    asm volatile("s_wait_asynccnt 0" ::: "memory");
    __syncthreads();
  }

  // ---- epilogue: out = x + gamma * (O / l) ----
  const float g = gamma[0];
  const float* xb2 = x   + (size_t)b * CCH * NPIX;
  float*       ob  = out + (size_t)b * CCH * NPIX;
  float inv[8];
#pragma unroll
  for (int r = 0; r < 8; ++r) inv[r] = 1.0f / lrow[r];

#pragma unroll
  for (int t = 0; t < 16; ++t) {
    const int c = t * 16 + laneLo;
    const size_t base = (size_t)c * NPIX + i0 + 8 * laneHi;
    float4 xv0 = *(const float4*)(xb2 + base);
    float4 xv1 = *(const float4*)(xb2 + base + 4);
    float o[8];
#pragma unroll
    for (int r = 0; r < 8; ++r) o[r] = acc[t][r] * inv[r];
    float4 r0, r1;
    r0.x = xv0.x + g * o[0]; r0.y = xv0.y + g * o[1];
    r0.z = xv0.z + g * o[2]; r0.w = xv0.w + g * o[3];
    r1.x = xv1.x + g * o[4]; r1.y = xv1.y + g * o[5];
    r1.z = xv1.z + g * o[6]; r1.w = xv1.w + g * o[7];
    *(float4*)(ob + base)     = r0;
    *(float4*)(ob + base + 4) = r1;
  }
}

// ---------------------------------------------------------------------------
extern "C" void kernel_launch(void* const* d_in, const int* in_sizes, int n_in,
                              void* d_out, int out_size, void* d_ws, size_t ws_size,
                              hipStream_t stream) {
  const float* x     = (const float*)d_in[0];
  const float* Wq    = (const float*)d_in[1];
  const float* bq    = (const float*)d_in[2];
  const float* Wk    = (const float*)d_in[3];
  const float* bk    = (const float*)d_in[4];
  const float* Wv    = (const float*)d_in[5];
  const float* bv    = (const float*)d_in[6];
  const float* gamma = (const float*)d_in[7];
  float* out = (float*)d_out;

  const size_t sz = (size_t)BATCH * NPIX * CCH;   // elements per f16 tensor
  half_t* xb = (half_t*)d_ws;                     // (B,N,C)
  half_t* Qt = xb + sz;                           // (B,N,C)
  half_t* Kt = Qt + sz;                           // (B,N,C)
  half_t* Vn = Kt + sz;                           // (B,C,N)
  half_t* Wh = Vn + sz;                           // 3 x (C,C) f16 weights

  hipLaunchKernelGGL(k_wconv, dim3(256), dim3(256), 0, stream, Wq, Wk, Wv, Wh);
  hipLaunchKernelGGL(k_transpose, dim3(BATCH * CCH), dim3(256), 0, stream, x, xb);
  hipLaunchKernelGGL(k_qkv, dim3(4096), dim3(256), 0, stream,
                     xb, Wh, bq, bk, bv, Qt, Kt, Vn);
  hipLaunchKernelGGL(k_attn, dim3(256), dim3(256), 0, stream,
                     Qt, Kt, Vn, x, gamma, out);
}
